// MultiHeadAttentionBlock_1632087573045
// MI455X (gfx1250) — matrix operand order repaired and checked by reading, not compiled, not run on hardware
//
#include <hip/hip_runtime.h>

// ---------------- CDNA5 (gfx1250) fused multi-head attention ----------------
// Pipeline: 3x WMMA projection GEMMs (fp32 -> bf16 head-split, Q pre-scaled by
// 1/sqrt(dk)), flash attention with v_wmma_f32_16x16x32_bf16 + online softmax
// (row-sum via WMMA against all-ones, K/V tiles double-buffered through LDS
// with GLOBAL_LOAD_ASYNC_TO_LDS_B128 / s_wait_asynccnt), final WMMA projection
// back to fp32. Scores never touch HBM.

typedef __bf16 bf16_t;
typedef __attribute__((ext_vector_type(16))) __bf16 v16bf;
typedef __attribute__((ext_vector_type(8)))  float  v8f;

union Frag {            // one 16x32 (A) or 32x16 (B) bf16 operand per wave
    v16bf v;
    uint4 u[2];
};

__device__ inline v8f wmma_bf16(const Frag& a, const Frag& b, v8f c) {
    return __builtin_amdgcn_wmma_f32_16x16x32_bf16(
        /*neg_a=*/false, a.v, /*neg_b=*/false, b.v,
        /*c_mod=*/(short)0, c, /*reuse_a=*/false, /*reuse_b=*/false);
}

// CDNA5 async global->LDS copy (16B per lane), tracked by ASYNCcnt.
__device__ inline void async_copy16(const void* gptr, void* lptr) {
    const unsigned lds_off = (unsigned)(uintptr_t)lptr;  // low 32b = LDS offset
    asm volatile("global_load_async_to_lds_b128 %0, %1, off"
                 :: "v"(lds_off), "v"(gptr) : "memory");
}
__device__ inline void wait_async0() {
    asm volatile("s_wait_asynccnt 0x0" ::: "memory");
}

constexpr int D_MODEL = 512;
constexpr int NHEAD   = 8;
constexpr int DK      = 64;
constexpr int S_LEN   = 4096;
constexpr int BATCH   = 2;
constexpr int MTOT    = BATCH * S_LEN;   // 8192

enum { OUT_HEADS = 0, OUT_HEADS_T = 1, OUT_FLAT_F32 = 2, OUT_HEADS_Q = 3 };

// ---------------------------------------------------------------------------
// Y[M,N] = X[M,K] @ W[N,K]^T + bias[N]   (M=8192, N=K=512)
// Block tile 128x64, 8 waves of 32x32, K-step 32 through LDS (bf16).
// ---------------------------------------------------------------------------
template <typename XT, int MODE>
__global__ __launch_bounds__(256)
void wmma_gemm_512(const XT* __restrict__ X, const float* __restrict__ W,
                   const float* __restrict__ bias, void* __restrict__ outp) {
    constexpr int BM = 128, BN = 64, BK = 32, LDA = 40, LDB = 40;
    __shared__ __align__(16) bf16_t As[BM * LDA];
    __shared__ __align__(16) bf16_t Bs[BN * LDB];

    const int tid  = threadIdx.x;
    const int bm0  = blockIdx.x * BM;
    const int bn0  = blockIdx.y * BN;
    const int wave = tid >> 5, lane = tid & 31;
    const int half = lane >> 4, lm = lane & 15;
    const int wy   = wave & 3, wx = wave >> 2;   // 4x2 wave grid

    v8f acc[2][2] = {};

    for (int k0 = 0; k0 < D_MODEL; k0 += BK) {
        const int col = (tid & 7) * 4;
        // ---- stage A tile (128x32), converting to bf16 ----
        #pragma unroll
        for (int p = 0; p < 4; ++p) {
            const int row = p * 32 + (tid >> 3);
            const XT* src = X + (size_t)(bm0 + row) * D_MODEL + k0 + col;
            bf16_t* dst = &As[row * LDA + col];
            if constexpr (sizeof(XT) == 4) {
                float4 s = *(const float4*)src;
                dst[0] = (bf16_t)s.x; dst[1] = (bf16_t)s.y;
                dst[2] = (bf16_t)s.z; dst[3] = (bf16_t)s.w;
            } else {
                *(uint2*)dst = *(const uint2*)src;   // already bf16
            }
        }
        // ---- stage B tile (64x32) from fp32 W ----
        #pragma unroll
        for (int p = 0; p < 2; ++p) {
            const int row = p * 32 + (tid >> 3);
            const float* src = W + (size_t)(bn0 + row) * D_MODEL + k0 + col;
            float4 s = *(const float4*)src;
            bf16_t* dst = &Bs[row * LDB + col];
            dst[0] = (bf16_t)s.x; dst[1] = (bf16_t)s.y;
            dst[2] = (bf16_t)s.z; dst[3] = (bf16_t)s.w;
        }
        if (k0 + BK < D_MODEL)   // hint next A tile into cache
            __builtin_prefetch(X + (size_t)(bm0 + (tid >> 3)) * D_MODEL + k0 + BK);
        __syncthreads();

        Frag a[2], b[2];
        #pragma unroll
        for (int i = 0; i < 2; ++i) {
            const int r = wy * 32 + i * 16 + lm;          // A: lane -> row
            a[i].u[0] = *(const uint4*)&As[r * LDA + half * 16];
            a[i].u[1] = *(const uint4*)&As[r * LDA + half * 16 + 8];
            const int c = wx * 32 + i * 16 + lm;          // B: lane -> col
            b[i].u[0] = *(const uint4*)&Bs[c * LDB + half * 16];
            b[i].u[1] = *(const uint4*)&Bs[c * LDB + half * 16 + 8];
        }
        #pragma unroll
        for (int i = 0; i < 2; ++i)
            #pragma unroll
            for (int j = 0; j < 2; ++j)
                acc[i][j] = wmma_bf16(a[i], b[j], acc[i][j]);
        __syncthreads();
    }

    // ---- epilogue: bias (+ optional 1/sqrt(dk) pre-scale) + scatter ----
    #pragma unroll
    for (int i = 0; i < 2; ++i)
        #pragma unroll
        for (int j = 0; j < 2; ++j) {
            const int n  = bn0 + wx * 32 + j * 16 + lm;
            const float bv = bias[n];
            #pragma unroll
            for (int r = 0; r < 8; ++r) {
                const int m = bm0 + wy * 32 + i * 16 + half * 8 + r;
                float v = acc[i][j][r] + bv;
                if constexpr (MODE == OUT_HEADS_Q) v *= 0.125f;  // exact in bf16
                if constexpr (MODE == OUT_FLAT_F32) {
                    ((float*)outp)[(size_t)m * D_MODEL + n] = v;
                } else {
                    const int bb = m / S_LEN, s = m % S_LEN;
                    const int hh = n / DK,   d = n % DK;
                    bf16_t* o = (bf16_t*)outp;
                    if constexpr (MODE == OUT_HEADS || MODE == OUT_HEADS_Q)
                        o[(((size_t)(bb * NHEAD + hh)) * S_LEN + s) * DK + d] = (bf16_t)v;
                    else // OUT_HEADS_T: [B,H,DK,S] so V columns are K-contiguous
                        o[(((size_t)(bb * NHEAD + hh)) * DK + d) * S_LEN + s] = (bf16_t)v;
                }
            }
        }
}

// ---------------------------------------------------------------------------
// Flash attention: block = 128 query rows (16 per wave), key tiles of 32,
// double-buffered async staging (issue tile i+1, compute tile i, wait, swap).
// Q (pre-scaled), K: bf16 [B,H,S,64]; V: bf16 [B,H,64,S]; ctx: bf16 [B,S,512].
// ---------------------------------------------------------------------------
__global__ __launch_bounds__(256)
void flash_attn(const bf16_t* __restrict__ Q, const bf16_t* __restrict__ Kb,
                const bf16_t* __restrict__ Vt, const int* __restrict__ mask,
                bf16_t* __restrict__ ctx) {
    constexpr int LDK = 72, LDV = 40, LDP = 40;
    __shared__ __align__(16) bf16_t Ks[2][32 * LDK];    // [buf][key][d]
    __shared__ __align__(16) bf16_t Vs[2][64 * LDV];    // [buf][d][key]
    __shared__ __align__(16) bf16_t Ps[8 * 16 * LDP];   // per-wave P transpose

    const int tid  = threadIdx.x;
    const int wave = tid >> 5, lane = tid & 31;
    const int half = lane >> 4, lm = lane & 15;
    const int bh = blockIdx.y, b = bh >> 3, h = bh & 7;
    const int q0 = blockIdx.x * 128 + wave * 16;

    // staging indices (constant per thread)
    const int skey = tid >> 3, sdseg = (tid & 7) * 8;   // K tile: 32x64
    const int sd   = tid >> 2, skseg = (tid & 3) * 8;   // V tile: 64x32
    const bf16_t* Kg = Kb + ((size_t)bh * S_LEN + skey) * DK + sdseg;
    const bf16_t* Vg = Vt + ((size_t)bh * DK + sd) * S_LEN + skseg;

    // Q block 16x64 -> two A operands (d 0..31, d 32..63), kept in VGPRs
    const bf16_t* Qp = Q + ((size_t)bh * S_LEN + q0) * DK;
    Frag aq0, aq1;
    aq0.u[0] = *(const uint4*)&Qp[lm * DK +      half * 16];
    aq0.u[1] = *(const uint4*)&Qp[lm * DK +  8 + half * 16];
    aq1.u[0] = *(const uint4*)&Qp[lm * DK + 32 + half * 16];
    aq1.u[1] = *(const uint4*)&Qp[lm * DK + 40 + half * 16];

    // all-ones B operand: row-sum of P via one extra WMMA
    Frag ones;
    ones.u[0] = make_uint4(0x3F803F80u, 0x3F803F80u, 0x3F803F80u, 0x3F803F80u);
    ones.u[1] = ones.u[0];

    float mrun[8], lrun[8];
    #pragma unroll
    for (int j = 0; j < 8; ++j) { mrun[j] = -1e30f; lrun[j] = 0.f; }
    v8f o[4] = {};

    // prologue: stage tile 0 into buffer 0
    bf16_t* ksC = Ks[0]; bf16_t* vsC = Vs[0];   // compute buffers
    bf16_t* ksN = Ks[1]; bf16_t* vsN = Vs[1];   // staging buffers
    async_copy16(Kg, &ksC[skey * LDK + sdseg]);
    async_copy16(Vg, &vsC[sd * LDV + skseg]);
    wait_async0();
    __syncthreads();

    for (int t = 0; t < S_LEN; t += 32) {
        // issue async staging of the NEXT tile into the alternate buffer
        if (t + 32 < S_LEN) {
            async_copy16(Kg + (size_t)(t + 32) * DK, &ksN[skey * LDK + sdseg]);
            async_copy16(Vg + (t + 32),              &vsN[sd * LDV + skseg]);
        }

        // S tile 16x32 = two 16x16 accumulators (already scaled via Q)
        v8f st[2];
        #pragma unroll
        for (int i = 0; i < 2; ++i) {
            Frag bk0, bk1;
            const int key = i * 16 + lm;
            bk0.u[0] = *(const uint4*)&ksC[key * LDK + half * 16];
            bk0.u[1] = *(const uint4*)&ksC[key * LDK + half * 16 + 8];
            bk1.u[0] = *(const uint4*)&ksC[key * LDK + 32 + half * 16];
            bk1.u[1] = *(const uint4*)&ksC[key * LDK + 32 + half * 16 + 8];
            v8f z = {};
            st[i] = wmma_bf16(aq0, bk0, z);
            st[i] = wmma_bf16(aq1, bk1, st[i]);
        }

        // mask + online softmax; row-max across the 16-lane half
        const int mk0 = mask[b * S_LEN + t + lm];
        const int mk1 = mask[b * S_LEN + t + 16 + lm];
        float mx[8], p0[8], p1[8];
        #pragma unroll
        for (int j = 0; j < 8; ++j) {
            float s0 = st[0][j]; if (mk0 == 0) s0 = -1e9f;
            float s1 = st[1][j]; if (mk1 == 0) s1 = -1e9f;
            st[0][j] = s0; st[1][j] = s1;
            mx[j] = fmaxf(s0, s1);
        }
        #pragma unroll
        for (int off = 1; off < 16; off <<= 1)
            #pragma unroll
            for (int j = 0; j < 8; ++j)
                mx[j] = fmaxf(mx[j], __shfl_xor(mx[j], off, 32));
        #pragma unroll
        for (int j = 0; j < 8; ++j) {
            const float mnew  = fmaxf(mrun[j], mx[j]);
            const float alpha = __expf(mrun[j] - mnew);
            p0[j] = __expf(st[0][j] - mnew);
            p1[j] = __expf(st[1][j] - mnew);
            mrun[j] = mnew;
            lrun[j] *= alpha;
            #pragma unroll
            for (int dt = 0; dt < 4; ++dt) o[dt][j] *= alpha;
        }

        // P (C-layout) -> A-operand layout via wave-private LDS transpose
        bf16_t* Pw = &Ps[wave * 16 * LDP];
        #pragma unroll
        for (int j = 0; j < 8; ++j) {
            Pw[(half * 8 + j) * LDP + lm]      = (bf16_t)p0[j];
            Pw[(half * 8 + j) * LDP + 16 + lm] = (bf16_t)p1[j];
        }
        Frag pa;
        pa.u[0] = *(const uint4*)&Pw[lm * LDP + half * 16];
        pa.u[1] = *(const uint4*)&Pw[lm * LDP + half * 16 + 8];

        // row-sum of (bf16-quantized) P via WMMA against ones
        v8f z = {};
        const v8f rsum = wmma_bf16(pa, ones, z);
        #pragma unroll
        for (int j = 0; j < 8; ++j) lrun[j] += rsum[j];

        // O[16x64] += P[16x32] @ V[32x64]
        #pragma unroll
        for (int dt = 0; dt < 4; ++dt) {
            Frag bv;
            bv.u[0] = *(const uint4*)&vsC[(dt * 16 + lm) * LDV + half * 16];
            bv.u[1] = *(const uint4*)&vsC[(dt * 16 + lm) * LDV + half * 16 + 8];
            o[dt] = wmma_bf16(pa, bv, o[dt]);
        }

        // drain this wave's async copies, then block-wide handoff; swap buffers
        wait_async0();
        __syncthreads();
        bf16_t* tk = ksC; ksC = ksN; ksN = tk;
        bf16_t* tv = vsC; vsC = vsN; vsN = tv;
    }

    // normalize and write merged-head ctx [B,S,512] bf16
    #pragma unroll
    for (int j = 0; j < 8; ++j) {
        const float inv = lrun[j] > 0.f ? 1.0f / lrun[j] : 0.f;
        const int s = q0 + half * 8 + j;
        #pragma unroll
        for (int dt = 0; dt < 4; ++dt) {
            const int dcol = h * DK + dt * 16 + lm;
            ctx[((size_t)b * S_LEN + s) * D_MODEL + dcol] = (bf16_t)(o[dt][j] * inv);
        }
    }
}

// ---------------------------------------------------------------------------
extern "C" void kernel_launch(void* const* d_in, const int* in_sizes, int n_in,
                              void* d_out, int out_size, void* d_ws, size_t ws_size,
                              hipStream_t stream) {
    const float* q    = (const float*)d_in[0];
    const float* k    = (const float*)d_in[1];
    const float* v    = (const float*)d_in[2];
    const int*   mask = (const int*)  d_in[3];
    const float* Wq   = (const float*)d_in[4];
    const float* bq   = (const float*)d_in[5];
    const float* Wk   = (const float*)d_in[6];
    const float* bk   = (const float*)d_in[7];
    const float* Wv   = (const float*)d_in[8];
    const float* bv   = (const float*)d_in[9];
    const float* Wo   = (const float*)d_in[10];
    const float* bo   = (const float*)d_in[11];

    char* ws = (char*)d_ws;
    const size_t seg = (size_t)BATCH * NHEAD * S_LEN * DK * sizeof(bf16_t); // 8 MB
    bf16_t* Qb  = (bf16_t*)(ws);
    bf16_t* Kb  = (bf16_t*)(ws + seg);
    bf16_t* Vt  = (bf16_t*)(ws + 2 * seg);
    bf16_t* Ctx = (bf16_t*)(ws + 3 * seg);

    dim3 gg(MTOT / 128, D_MODEL / 64);   // 64 x 8
    wmma_gemm_512<float,  OUT_HEADS_Q><<<gg, 256, 0, stream>>>(q, Wq, bq, Qb);
    wmma_gemm_512<float,  OUT_HEADS>  <<<gg, 256, 0, stream>>>(k, Wk, bk, Kb);
    wmma_gemm_512<float,  OUT_HEADS_T><<<gg, 256, 0, stream>>>(v, Wv, bv, Vt);

    flash_attn<<<dim3(S_LEN / 128, BATCH * NHEAD), 256, 0, stream>>>(Qb, Kb, Vt, mask, Ctx);

    wmma_gemm_512<bf16_t, OUT_FLAT_F32><<<gg, 256, 0, stream>>>(Ctx, Wo, bo, (float*)d_out);
}
